// ChannelAttention_50173807952003
// MI455X (gfx1250) — matrix operand order repaired and checked by reading, not compiled, not run on hardware
//
#include <hip/hip_runtime.h>
#include <hip/hip_bf16.h>
#include <stdint.h>

// ---------------------------------------------------------------------------
// Channel attention, MI455X (gfx1250, wave32, WMMA).
//   energy = x x^T (bf16 WMMA, f32 acc) -> softmax over axis=1 (diag cancels)
//   out    = attn x (bf16 WMMA, f32 acc), final = gamma*out + inputs
// Compute-bound (~257 FLOP/byte). Each wave computes a 32x64 output strip
// (2 A-tiles x 4 B-tiles, 8 accumulators -> 1.5 b128 loads per WMMA); the 8
// waves of a block share the same 32 A-rows (WGP$ reuse) and together cover a
// full 512-wide B panel, halving L2 read traffic vs 16x64 strips.
// ---------------------------------------------------------------------------

typedef __attribute__((ext_vector_type(16))) __bf16 v16bf;
typedef __attribute__((ext_vector_type(8)))  __bf16 v8bf;
typedef __attribute__((ext_vector_type(8)))  float  v8f;

#define BATCH 16
#define CCH   512
#define NPIX  4096   // H*W = 64*64

static __device__ __forceinline__ uint16_t f32_to_bf16_rne(float f) {
    union { float f; uint32_t u; } v; v.f = f;
    uint32_t u = v.u;
    u += 0x7FFFu + ((u >> 16) & 1u);   // round-to-nearest-even
    return (uint16_t)(u >> 16);
}

// WGP-scope prefetch (scope 0 pulls into all cache levels on miss; the
// __builtin_prefetch locality arg only reaches SCOPE_SE/SYS which skip WGP$).
static __device__ __forceinline__ void prefetch_wgp(const void* p) {
    asm volatile("global_prefetch_b8 %0, off" :: "v"(p) : "memory");
}

// ---------------------------------------------------------------------------
// Kernel 1: f32 -> bf16 conversion, writing both row-major xb (C x N per
// batch) and transposed xbT (N x C per batch) via an LDS-tiled transpose.
// ---------------------------------------------------------------------------
__global__ void ca_convert_transpose(const float* __restrict__ x,
                                     uint16_t* __restrict__ xb,
                                     uint16_t* __restrict__ xbT) {
    __shared__ uint16_t tile[64][65];       // +1 pad: avoid bank conflicts
    const int b  = blockIdx.z;
    const int c0 = blockIdx.y * 64;
    const int n0 = blockIdx.x * 64;
    const int tx = threadIdx.x & 63;
    const int ty = threadIdx.x >> 6;        // 0..3
    const float*    Xf = x   + (size_t)b * CCH * NPIX;
    uint16_t*       Xb = xb  + (size_t)b * CCH * NPIX;
    uint16_t*       Xt = xbT + (size_t)b * NPIX * CCH;
    #pragma unroll
    for (int r = 0; r < 64; r += 4) {
        const int row = r + ty;
        const float f = Xf[(size_t)(c0 + row) * NPIX + (n0 + tx)];
        const uint16_t h = f32_to_bf16_rne(f);
        Xb[(size_t)(c0 + row) * NPIX + (n0 + tx)] = h;
        tile[row][tx] = h;
    }
    __syncthreads();
    #pragma unroll
    for (int r = 0; r < 64; r += 4) {
        const int row = r + ty;
        Xt[(size_t)(n0 + row) * CCH + (c0 + tx)] = tile[tx][row];
    }
}

// ---------------------------------------------------------------------------
// Kernel 2: energy[b,i,j] = sum_n xb[b,i,n]*xb[b,j,n].
// Block = 32(i) x 512(j) strip: wave w handles j in [64w, 64w+64).
// Per wave: 2 A-tiles x 4 B-tiles, 8 WMMAs / 12 b128 loads per K-step of 32.
// Both operands are contiguous row segments of xb (symmetric GEMM).
// ---------------------------------------------------------------------------
__global__ void ca_energy_wmma(const uint16_t* __restrict__ xb,
                               float* __restrict__ energy) {
    const int lane = threadIdx.x & 31;
    const int wave = threadIdx.x >> 5;            // 0..7
    const int b    = blockIdx.y;
    const int is0  = blockIdx.x * 32;             // i-superstrip (C/32 = 16)
    const int j0   = wave * 64;                   // j-supertile per wave
    const int hf = lane >> 4;                     // which half of the wave
    const int lm = lane & 15;

    const __bf16* X = (const __bf16*)(xb + (size_t)b * CCH * NPIX);
    const v8bf* arow[2];
    #pragma unroll
    for (int u = 0; u < 2; ++u)
        arow[u] = (const v8bf*)(X + (size_t)(is0 + 16 * u + lm) * NPIX);
    const v8bf* brow[4];
    #pragma unroll
    for (int t = 0; t < 4; ++t)
        brow[t] = (const v8bf*)(X + (size_t)(j0 + 16 * t + lm) * NPIX);

    union V16 { v16bf v; v8bf h[2]; };
    v8f acc[2][4] = {{v8f{}, v8f{}, v8f{}, v8f{}},
                     {v8f{}, v8f{}, v8f{}, v8f{}}};
    for (int k = 0; k < NPIX; k += 32) {
        const int kb = k >> 3;                    // offset in 8-elem units
        // prefetch ~4 iterations ahead into all cache levels (WGP scope)
        prefetch_wgp((const void*)(arow[0] + kb + 16));
        prefetch_wgp((const void*)(arow[1] + kb + 16));
        prefetch_wgp((const void*)(brow[0] + kb + 16));
        prefetch_wgp((const void*)(brow[1] + kb + 16));
        prefetch_wgp((const void*)(brow[2] + kb + 16));
        prefetch_wgp((const void*)(brow[3] + kb + 16));
        V16 a[2];
        #pragma unroll
        for (int u = 0; u < 2; ++u) {
            a[u].h[0] = arow[u][kb + hf];         // K = k + 8*hf     .. +8
            a[u].h[1] = arow[u][kb + 2 + hf];     // K = k +16+ 8*hf  .. +8
        }
        #pragma unroll
        for (int t = 0; t < 4; ++t) {
            V16 bm;
            bm.h[0] = brow[t][kb + hf];           // K = k + 8*hf     .. +8
            bm.h[1] = brow[t][kb + 2 + hf];       // K = k +16+ 8*hf  .. +8
            #pragma unroll
            for (int u = 0; u < 2; ++u)
                acc[u][t] = __builtin_amdgcn_wmma_f32_16x16x32_bf16(
                    false, a[u].v, false, bm.v, (short)0, acc[u][t],
                    false, false);
        }
    }
    float* E = energy + (size_t)b * CCH * CCH;
    #pragma unroll
    for (int u = 0; u < 2; ++u)
        #pragma unroll
        for (int t = 0; t < 4; ++t)
            #pragma unroll
            for (int r = 0; r < 8; ++r)           // C/D layout: M=r+8*hf, N=lm
                E[(size_t)(is0 + 16 * u + r + hf * 8) * CCH +
                  (j0 + 16 * t + lm)] = acc[u][t][r];
}

// ---------------------------------------------------------------------------
// Kernel 3: per-column (axis=1) softmax of (diag - energy); diag cancels:
//   m[j] = min_i e[i,j];  attn[i,j] = exp(m[j]-e[i,j]) / sum_i exp(m[j]-e[i,j])
// One thread per column j; coalesced across threads at each i. Writes bf16.
// ---------------------------------------------------------------------------
__global__ void ca_softmax(const float* __restrict__ energy,
                           uint16_t* __restrict__ attnb) {
    const int b = blockIdx.y;
    const int j = blockIdx.x * blockDim.x + threadIdx.x;   // 0..511
    const float* E = energy + (size_t)b * CCH * CCH;
    float mn = 3.402823466e38f;
    for (int i = 0; i < CCH; ++i)
        mn = fminf(mn, E[(size_t)i * CCH + j]);
    float s = 0.0f;
    for (int i = 0; i < CCH; ++i)
        s += __expf(mn - E[(size_t)i * CCH + j]);
    const float inv = 1.0f / s;
    uint16_t* A = attnb + (size_t)b * CCH * CCH;
    for (int i = 0; i < CCH; ++i) {
        const float a = __expf(mn - E[(size_t)i * CCH + j]) * inv;
        A[(size_t)i * CCH + j] = f32_to_bf16_rne(a);
    }
}

// ---------------------------------------------------------------------------
// Kernel 4: out[b,i,n] = sum_j attn[b,i,j]*x[b,j,n];  y = gamma*out + x.
// A = attn (row-major, K=j contiguous); B from xbT (rows are n, K=j contig).
// Block = 32(i) x 512(n); wave w handles n in [n_base + 64w, +64).
// ---------------------------------------------------------------------------
__global__ void ca_out_wmma(const uint16_t* __restrict__ attnb,
                            const uint16_t* __restrict__ xbT,
                            const float* __restrict__ x,
                            const float* __restrict__ gamma,
                            float* __restrict__ out) {
    const int lane = threadIdx.x & 31;
    const int wave = threadIdx.x >> 5;
    const int b    = blockIdx.y;
    const int is0  = (blockIdx.x >> 3) * 32;          // i-superstrip (16)
    const int n0   = (blockIdx.x & 7) * 512 + wave * 64;  // n-supertile
    const int hf = lane >> 4;
    const int lm = lane & 15;

    const __bf16* A  = (const __bf16*)(attnb + (size_t)b * CCH * CCH);
    const __bf16* Bt = (const __bf16*)(xbT   + (size_t)b * NPIX * CCH);
    const v8bf* arow[2];
    #pragma unroll
    for (int u = 0; u < 2; ++u)
        arow[u] = (const v8bf*)(A + (size_t)(is0 + 16 * u + lm) * CCH);
    const v8bf* brow[4];
    #pragma unroll
    for (int t = 0; t < 4; ++t)
        brow[t] = (const v8bf*)(Bt + (size_t)(n0 + 16 * t + lm) * CCH);

    union V16 { v16bf v; v8bf h[2]; };
    v8f acc[2][4] = {{v8f{}, v8f{}, v8f{}, v8f{}},
                     {v8f{}, v8f{}, v8f{}, v8f{}}};
    #pragma unroll 2
    for (int k = 0; k < CCH; k += 32) {               // K = j, 16 iterations
        const int kb = k >> 3;
        V16 a[2];
        #pragma unroll
        for (int u = 0; u < 2; ++u) {
            a[u].h[0] = arow[u][kb + hf];
            a[u].h[1] = arow[u][kb + 2 + hf];
        }
        #pragma unroll
        for (int t = 0; t < 4; ++t) {
            V16 bm;
            bm.h[0] = brow[t][kb + hf];
            bm.h[1] = brow[t][kb + 2 + hf];
            #pragma unroll
            for (int u = 0; u < 2; ++u)
                acc[u][t] = __builtin_amdgcn_wmma_f32_16x16x32_bf16(
                    false, a[u].v, false, bm.v, (short)0, acc[u][t],
                    false, false);
        }
    }
    const float g = gamma[0];
    const float* Xin = x   + (size_t)b * CCH * NPIX;
    float*       Y   = out + (size_t)b * CCH * NPIX;
    #pragma unroll
    for (int u = 0; u < 2; ++u)
        #pragma unroll
        for (int t = 0; t < 4; ++t)
            #pragma unroll
            for (int r = 0; r < 8; ++r) {
                const size_t idx =
                    (size_t)(is0 + 16 * u + r + hf * 8) * NPIX +
                    (n0 + 16 * t + lm);
                Y[idx] = g * acc[u][t][r] + Xin[idx];
            }
}

// ---------------------------------------------------------------------------
extern "C" void kernel_launch(void* const* d_in, const int* in_sizes, int n_in,
                              void* d_out, int out_size, void* d_ws, size_t ws_size,
                              hipStream_t stream) {
    const float* x     = (const float*)d_in[0];   // (16,512,64,64) f32
    const float* gamma = (const float*)d_in[1];   // (1,) f32
    float* out = (float*)d_out;

    // Workspace carving (~152 MiB total):
    //   xb   : bf16 (B,C,N)   64 MiB
    //   xbT  : bf16 (B,N,C)   64 MiB
    //   ener : f32  (B,C,C)   16 MiB
    //   attn : bf16 (B,C,C)    8 MiB
    char* ws = (char*)d_ws;
    const size_t xb_bytes = (size_t)BATCH * CCH * NPIX * 2;
    uint16_t* xb   = (uint16_t*)(ws);
    uint16_t* xbT  = (uint16_t*)(ws + xb_bytes);
    float*    ener = (float*)   (ws + 2 * xb_bytes);
    uint16_t* attn = (uint16_t*)(ws + 2 * xb_bytes + (size_t)BATCH * CCH * CCH * 4);

    // 1) convert + transpose: tiles of 64x64, grid (N/64, C/64, B)
    ca_convert_transpose<<<dim3(NPIX / 64, CCH / 64, BATCH), 256, 0, stream>>>(
        x, xb, xbT);

    // 2) energy GEMM: 16 blocks per batch, each a 32x512 strip
    ca_energy_wmma<<<dim3(16, BATCH), 256, 0, stream>>>(xb, ener);

    // 3) softmax along axis=1 (one thread per column)
    ca_softmax<<<dim3(CCH / 256, BATCH), 256, 0, stream>>>(ener, attn);

    // 4) out GEMM + epilogue: 16 i-strips x 8 n-groups = 128 blocks per batch
    ca_out_wmma<<<dim3(128, BATCH), 256, 0, stream>>>(attn, xbT, x, gamma, out);
}
